// BiMambaBlock_47115791237833
// MI455X (gfx1250) — compile-verified, matrix operand order checked by reading, not run on hardware
//
#include <hip/hip_runtime.h>
#include <hip/hip_bf16.h>

// ---------------- problem dims ----------------
#define BATCH    8
#define L_SEQ    3072
#define D_MODEL  64
#define D_INNER  128
#define D_STATE  16
#define DT_RANK  4
#define D_CONV   4
#define NTOK     (BATCH * L_SEQ)        // 24576
#define XDBL_ST  48                     // padded stride for x_dbl (36 -> 48)
#define CHUNK    128
#define NCHUNK   (L_SEQ / CHUNK)        // 24

typedef __attribute__((ext_vector_type(16))) _Float16 v16h;
typedef __attribute__((ext_vector_type(8)))  float    v8f;

// chunk-state layout: [b][chunk][s][c]  (c fastest -> coalesced)
#define CBASE(b, ch) (((size_t)((b) * NCHUNK + (ch)) * D_STATE) * D_INNER)

// ============================================================
// K1: fused RMSNorm + in_proj GEMM via WMMA f16->f32
// ============================================================
__global__ __launch_bounds__(128) void k1_norm_inproj(
    const float* __restrict__ x, const float* __restrict__ norm_w,
    const float* __restrict__ in_proj, float* __restrict__ xcv,
    float* __restrict__ zbuf, int dir)
{
    __shared__ _Float16 As[16][72];   // 16 tokens x 64 feats, padded
    const int tid = threadIdx.x;
    const int tk0 = blockIdx.x * 16;

    // threads 0..63: RMSNorm, 4 threads per token, shuffle-reduced (wave32)
    if (tid < 64) {
        int tok = tid >> 2, part = tid & 3;
        int g = tk0 + tok;
        int b = g / L_SEQ, t = g % L_SEQ;
        int tp = dir ? (L_SEQ - 1 - t) : t;
        const float* row = x + (size_t)(b * L_SEQ + tp) * D_MODEL + part * 16;
        float4 r0 = ((const float4*)row)[0];
        float4 r1 = ((const float4*)row)[1];
        float4 r2 = ((const float4*)row)[2];
        float4 r3 = ((const float4*)row)[3];
        float ss = r0.x*r0.x + r0.y*r0.y + r0.z*r0.z + r0.w*r0.w
                 + r1.x*r1.x + r1.y*r1.y + r1.z*r1.z + r1.w*r1.w
                 + r2.x*r2.x + r2.y*r2.y + r2.z*r2.z + r2.w*r2.w
                 + r3.x*r3.x + r3.y*r3.y + r3.z*r3.z + r3.w*r3.w;
        ss += __shfl_xor(ss, 1, 32);
        ss += __shfl_xor(ss, 2, 32);
        float sc = rsqrtf(ss * (1.0f / D_MODEL) + 1e-5f);
        float rv[16] = { r0.x,r0.y,r0.z,r0.w, r1.x,r1.y,r1.z,r1.w,
                         r2.x,r2.y,r2.z,r2.w, r3.x,r3.y,r3.z,r3.w };
        #pragma unroll
        for (int k = 0; k < 16; ++k) {
            int kk = part * 16 + k;
            As[tok][kk] = (_Float16)(rv[k] * sc * norm_w[kk]);
        }
    }
    __syncthreads();

    const int wave = tid >> 5, lane = tid & 31;
    const int mrow = lane & 15, hf = lane >> 4;

    #pragma unroll
    for (int tt = 0; tt < 4; ++tt) {
        const int ntile = wave + 4 * tt;                 // 0..15 -> 256 cols
        const int ncol  = ntile * 16 + (lane & 15);
        v8f acc = {};
        #pragma unroll
        for (int kk = 0; kk < 2; ++kk) {
            v16h a, bf;
            #pragma unroll
            for (int i = 0; i < 16; ++i) {
                int v = i >> 1, sub = i & 1;
                int k = ((v >> 2) << 4) + hf * 8 + ((v & 3) << 1) + sub;
                a[i]  = As[mrow][kk * 32 + k];
                bf[i] = (_Float16)in_proj[(size_t)ncol * D_MODEL + kk * 32 + k];
            }
            acc = __builtin_amdgcn_wmma_f32_16x16x32_f16(
                false, a, false, bf, (short)0, acc, false, false);
        }
        // wave-uniform destination select (ntile<8 -> xcv, else zbuf)
        float* obase = (ntile < 8)
            ? (xcv  + (size_t)tk0 * D_INNER + ntile * 16)
            : (zbuf + (size_t)tk0 * D_INNER + (ntile - 8) * 16);
        #pragma unroll
        for (int v = 0; v < 8; ++v)
            obase[(size_t)(v + 8 * hf) * D_INNER + (lane & 15)] = acc[v];
    }
}

// ============================================================
// K2: causal depthwise conv (4-tap, logical time) + SiLU
// ============================================================
__global__ __launch_bounds__(256) void k2_conv_silu(
    const float* __restrict__ xcv, const float* __restrict__ conv_w,
    const float* __restrict__ conv_b, float* __restrict__ xact)
{
    size_t i = (size_t)blockIdx.x * blockDim.x + threadIdx.x;   // NTOK*128
    int c = (int)(i & (D_INNER - 1));
    int g = (int)(i >> 7);
    int b = g / L_SEQ, t = g % L_SEQ;
    float s = conv_b[c];
    #pragma unroll
    for (int k = 0; k < D_CONV; ++k) {
        int tt = t - (D_CONV - 1) + k;
        if (tt >= 0)
            s += conv_w[c * D_CONV + k] * xcv[(size_t)(b * L_SEQ + tt) * D_INNER + c];
    }
    xact[i] = s / (1.f + __expf(-s));   // SiLU
}

// ============================================================
// K3: x_proj GEMM (N=36 padded to 48) via WMMA; branchless stores
// ============================================================
__global__ __launch_bounds__(96) void k3_xproj(
    const float* __restrict__ xact, const float* __restrict__ x_proj,
    float* __restrict__ xdbl)
{
    const int tid = threadIdx.x, wave = tid >> 5, lane = tid & 31;
    const int tk0 = blockIdx.x * 16;
    const int m = lane & 15, hf = lane >> 4;
    const int ncol = wave * 16 + (lane & 15);   // 0..47
    v8f acc = {};
    #pragma unroll
    for (int kk = 0; kk < 4; ++kk) {
        v16h a, bf;
        #pragma unroll
        for (int i = 0; i < 16; ++i) {
            int v = i >> 1, sub = i & 1;
            int k = ((v >> 2) << 4) + hf * 8 + ((v & 3) << 1) + sub;
            int kg = kk * 32 + k;
            a[i]  = (_Float16)xact[(size_t)(tk0 + m) * D_INNER + kg];
            bf[i] = (ncol < 36) ? (_Float16)x_proj[(size_t)ncol * D_INNER + kg]
                                : (_Float16)0.f;
        }
        acc = __builtin_amdgcn_wmma_f32_16x16x32_f16(
            false, a, false, bf, (short)0, acc, false, false);
    }
    // cols 36..47 are pad, never read -> store unconditionally
    #pragma unroll
    for (int v = 0; v < 8; ++v)
        xdbl[(size_t)(tk0 + v + 8 * hf) * XDBL_ST + ncol] = acc[v];
}

// ============================================================
// K4: delta = softplus(dt @ dt_w^T + dt_b)   (tiny K=4 -> VALU)
// ============================================================
__global__ __launch_bounds__(256) void k4_delta(
    const float* __restrict__ xdbl, const float* __restrict__ dt_w,
    const float* __restrict__ dt_b, float* __restrict__ delta)
{
    size_t i = (size_t)blockIdx.x * blockDim.x + threadIdx.x;
    int c = (int)(i & (D_INNER - 1));
    int g = (int)(i >> 7);
    float4 dt4 = *(const float4*)(xdbl + (size_t)g * XDBL_ST);
    float4 w4  = *(const float4*)(dt_w + c * DT_RANK);
    float acc = dt_b[c] + dt4.x * w4.x + dt4.y * w4.y + dt4.z * w4.z + dt4.w * w4.w;
    delta[i] = (acc > 20.f) ? acc : log1pf(__expf(acc));
}

// ============================================================
// K5: scan pass A — per-chunk decay product P and partial q
//     thread order: c fastest -> coalesced delta/xact loads
// ============================================================
__global__ __launch_bounds__(256) void k5_scan_partial(
    const float* __restrict__ delta, const float* __restrict__ xact,
    const float* __restrict__ xdbl, const float* __restrict__ A_log,
    float* __restrict__ chunkP, float* __restrict__ chunkQ)
{
    int i  = blockIdx.x * blockDim.x + threadIdx.x;   // B*NCHUNK*128
    int c  = i % D_INNER;
    int ch = (i / D_INNER) % NCHUNK;
    int b  = i / (D_INNER * NCHUNK);
    float A[D_STATE], h[D_STATE], P[D_STATE];
    #pragma unroll
    for (int s = 0; s < D_STATE; ++s) {
        A[s] = -__expf(A_log[c * D_STATE + s]);
        h[s] = 0.f; P[s] = 1.f;
    }
    size_t g0 = (size_t)b * L_SEQ + (size_t)ch * CHUNK;
    for (int t = 0; t < CHUNK; ++t) {
        size_t g = g0 + t;
        if (t + 1 < CHUNK) {
            __builtin_prefetch(&delta[(g + 1) * D_INNER + c], 0, 0);
            __builtin_prefetch(&xdbl[(g + 1) * XDBL_ST], 0, 0);
        }
        float d  = delta[g * D_INNER + c];
        float xa = xact [g * D_INNER + c];
        float dx = d * xa;
        const float4* B4 = (const float4*)(xdbl + g * XDBL_ST + DT_RANK);
        float Bs[D_STATE];
        *(float4*)&Bs[0]  = B4[0]; *(float4*)&Bs[4]  = B4[1];
        *(float4*)&Bs[8]  = B4[2]; *(float4*)&Bs[12] = B4[3];
        #pragma unroll
        for (int s = 0; s < D_STATE; ++s) {
            float dA = __expf(d * A[s]);
            h[s] = dA * h[s] + dx * Bs[s];
            P[s] *= dA;
        }
    }
    size_t o = CBASE(b, ch) + c;
    #pragma unroll
    for (int s = 0; s < D_STATE; ++s) {
        chunkP[o + (size_t)s * D_INNER] = P[s];
        chunkQ[o + (size_t)s * D_INNER] = h[s];
    }
}

// ============================================================
// K6: scan pass B — stitch chunk boundaries (24 sequential steps)
// ============================================================
__global__ __launch_bounds__(256) void k6_scan_stitch(
    const float* __restrict__ chunkP, const float* __restrict__ chunkQ,
    float* __restrict__ h0)
{
    int i = blockIdx.x * blockDim.x + threadIdx.x;   // B*16*128
    int c = i % D_INNER;
    int s = (i / D_INNER) % D_STATE;
    int b = i / (D_INNER * D_STATE);
    float hc = 0.f;
    for (int ch = 0; ch < NCHUNK; ++ch) {
        size_t o = CBASE(b, ch) + (size_t)s * D_INNER + c;
        h0[o] = hc;
        hc = chunkP[o] * hc + chunkQ[o];
    }
}

// ============================================================
// K7: scan pass C — rerun chunks from corrected h0, emit
//     y = (sum_s h*C + D*x) * SiLU(z)   into ygate
// ============================================================
__global__ __launch_bounds__(256) void k7_scan_emit(
    const float* __restrict__ delta, const float* __restrict__ xact,
    const float* __restrict__ xdbl, const float* __restrict__ A_log,
    const float* __restrict__ Dp, const float* __restrict__ zbuf,
    const float* __restrict__ h0, float* __restrict__ ygate)
{
    int i  = blockIdx.x * blockDim.x + threadIdx.x;
    int c  = i % D_INNER;
    int ch = (i / D_INNER) % NCHUNK;
    int b  = i / (D_INNER * NCHUNK);
    float A[D_STATE], h[D_STATE];
    size_t o = CBASE(b, ch) + c;
    #pragma unroll
    for (int s = 0; s < D_STATE; ++s) {
        A[s] = -__expf(A_log[c * D_STATE + s]);
        h[s] = h0[o + (size_t)s * D_INNER];
    }
    float Dc = Dp[c];
    size_t g0 = (size_t)b * L_SEQ + (size_t)ch * CHUNK;
    for (int t = 0; t < CHUNK; ++t) {
        size_t g = g0 + t;
        if (t + 1 < CHUNK) {
            __builtin_prefetch(&delta[(g + 1) * D_INNER + c], 0, 0);
            __builtin_prefetch(&xdbl[(g + 1) * XDBL_ST], 0, 0);
        }
        float d  = delta[g * D_INNER + c];
        float xa = xact [g * D_INNER + c];
        float dx = d * xa;
        const float4* B4 = (const float4*)(xdbl + g * XDBL_ST + DT_RANK);
        const float4* C4 = (const float4*)(xdbl + g * XDBL_ST + DT_RANK + D_STATE);
        float Bs[D_STATE], Cs[D_STATE];
        *(float4*)&Bs[0]  = B4[0]; *(float4*)&Bs[4]  = B4[1];
        *(float4*)&Bs[8]  = B4[2]; *(float4*)&Bs[12] = B4[3];
        *(float4*)&Cs[0]  = C4[0]; *(float4*)&Cs[4]  = C4[1];
        *(float4*)&Cs[8]  = C4[2]; *(float4*)&Cs[12] = C4[3];
        float y = 0.f;
        #pragma unroll
        for (int s = 0; s < D_STATE; ++s) {
            float dA = __expf(d * A[s]);
            h[s] = dA * h[s] + dx * Bs[s];
            y += h[s] * Cs[s];
        }
        float z = zbuf[g * D_INNER + c];
        float sz = z / (1.f + __expf(-z));
        ygate[g * D_INNER + c] = (y + Dc * xa) * sz;
    }
}

// ============================================================
// K8: out_proj GEMM via WMMA + residual; de-flip for bwd
// ============================================================
__global__ __launch_bounds__(128) void k8_outproj(
    const float* __restrict__ ygate, const float* __restrict__ out_proj,
    const float* __restrict__ x, float* __restrict__ out, int dir)
{
    const int tid = threadIdx.x, wave = tid >> 5, lane = tid & 31;
    const int tk0 = blockIdx.x * 16;
    const int m = lane & 15, hf = lane >> 4;
    const int ncol = wave * 16 + (lane & 15);   // 0..63
    v8f acc = {};
    #pragma unroll
    for (int kk = 0; kk < 4; ++kk) {
        v16h a, bf;
        #pragma unroll
        for (int i = 0; i < 16; ++i) {
            int v = i >> 1, sub = i & 1;
            int k = ((v >> 2) << 4) + hf * 8 + ((v & 3) << 1) + sub;
            int kg = kk * 32 + k;
            a[i]  = (_Float16)ygate[(size_t)(tk0 + m) * D_INNER + kg];
            bf[i] = (_Float16)out_proj[(size_t)ncol * D_INNER + kg];
        }
        acc = __builtin_amdgcn_wmma_f32_16x16x32_f16(
            false, a, false, bf, (short)0, acc, false, false);
    }
    #pragma unroll
    for (int v = 0; v < 8; ++v) {
        int g = tk0 + v + 8 * hf;
        int b = g / L_SEQ, t = g % L_SEQ;
        int tp = dir ? (L_SEQ - 1 - t) : t;
        size_t pt = (size_t)b * L_SEQ + tp;
        out[pt * (2 * D_MODEL) + dir * D_MODEL + ncol] =
            acc[v] + x[pt * D_MODEL + ncol];
    }
}

// ============================================================
extern "C" void kernel_launch(void* const* d_in, const int* in_sizes, int n_in,
                              void* d_out, int out_size, void* d_ws, size_t ws_size,
                              hipStream_t stream) {
    (void)in_sizes; (void)n_in; (void)out_size; (void)ws_size;
    const float* x = (const float*)d_in[0];
    float* out = (float*)d_out;

    float* ws = (float*)d_ws;
    const size_t NC128 = (size_t)NTOK * D_INNER;
    float* xcv    = ws;                       // reused as ygate by K7/K8
    float* zbuf   = xcv   + NC128;
    float* xact   = zbuf  + NC128;
    float* xdbl   = xact  + NC128;            // NTOK * 48
    float* delta  = xdbl  + (size_t)NTOK * XDBL_ST;
    float* chunkP = delta + NC128;            // B*NCHUNK*16*128
    const size_t CSZ = (size_t)BATCH * NCHUNK * D_STATE * D_INNER;
    float* chunkQ = chunkP + CSZ;
    float* h0     = chunkQ + CSZ;

    for (int dir = 0; dir < 2; ++dir) {
        const int pb = 1 + dir * 10;
        const float* norm_w   = (const float*)d_in[pb + 0];
        const float* in_proj  = (const float*)d_in[pb + 1];
        const float* conv_w   = (const float*)d_in[pb + 2];
        const float* conv_b   = (const float*)d_in[pb + 3];
        const float* x_proj   = (const float*)d_in[pb + 4];
        const float* dt_w     = (const float*)d_in[pb + 5];
        const float* dt_b     = (const float*)d_in[pb + 6];
        const float* A_log    = (const float*)d_in[pb + 7];
        const float* Dp       = (const float*)d_in[pb + 8];
        const float* out_proj = (const float*)d_in[pb + 9];

        k1_norm_inproj<<<NTOK / 16, 128, 0, stream>>>(x, norm_w, in_proj, xcv, zbuf, dir);
        k2_conv_silu  <<<(NC128) / 256, 256, 0, stream>>>(xcv, conv_w, conv_b, xact);
        k3_xproj      <<<NTOK / 16, 96, 0, stream>>>(xact, x_proj, xdbl);
        k4_delta      <<<(NC128) / 256, 256, 0, stream>>>(xdbl, dt_w, dt_b, delta);
        k5_scan_partial<<<(BATCH * D_INNER * NCHUNK) / 256, 256, 0, stream>>>(
            delta, xact, xdbl, A_log, chunkP, chunkQ);
        k6_scan_stitch<<<(BATCH * D_INNER * D_STATE) / 256, 256, 0, stream>>>(
            chunkP, chunkQ, h0);
        k7_scan_emit  <<<(BATCH * D_INNER * NCHUNK) / 256, 256, 0, stream>>>(
            delta, xact, xdbl, A_log, Dp, zbuf, h0, /*ygate=*/xcv);
        k8_outproj    <<<NTOK / 16, 128, 0, stream>>>(xcv, out_proj, x, out, dir);
    }
}
